// GateLoop_37821482009321
// MI455X (gfx1250) — matrix operand order repaired and checked by reading, not compiled, not run on hardware
//
#include <hip/hip_runtime.h>
#include <hip/hip_bf16.h>

typedef __bf16 bf16;
typedef bf16  v16bf  __attribute__((ext_vector_type(16)));
typedef bf16  bf16x8 __attribute__((ext_vector_type(8)));
typedef float v8f    __attribute__((ext_vector_type(8)));

// 128-bit chunk type + address-space-qualified pointers for the async builtin.
// (Prototype per hipcc diagnostic: params are v4i ptrs in AS1 / AS3.)
typedef int gl_v4i __attribute__((vector_size(16)));
typedef __attribute__((address_space(1))) gl_v4i gl_gv4i;
typedef __attribute__((address_space(3))) gl_v4i gl_lv4i;

#if defined(__has_builtin)
#  if __has_builtin(__builtin_amdgcn_global_load_async_to_lds_b128)
#    define GL_ASYNC 1
#  endif
#endif
#ifndef GL_ASYNC
#  define GL_ASYNC 0
#endif

__device__ __forceinline__ void gl_wait_async() {
#if defined(__has_builtin)
#  if __has_builtin(__builtin_amdgcn_s_wait_asynccnt)
  __builtin_amdgcn_s_wait_asynccnt(0);
#  else
  asm volatile("s_wait_asynccnt 0" ::: "memory");
#  endif
#else
  asm volatile("s_wait_asynccnt 0" ::: "memory");
#endif
}

#define BB 4
#define NN 4096
#define DD 1024
#define BN (BB * NN)        // 16384 rows
#define NC6 (6 * DD)        // 6144 concat proj cols
#define CHUNK 64
#define NCHUNK (NN / CHUNK) // 64

// ---------------------------------------------------------------------------
// Kernel 1: RMSNorm + cast to bf16.
// ---------------------------------------------------------------------------
__global__ __launch_bounds__(256) void gl_rmsnorm_kernel(
    const float* __restrict__ x, const float* __restrict__ gamma,
    bf16* __restrict__ xn) {
  __shared__ float red[8];
  const int row = blockIdx.x;
  const float* xr = x + (size_t)row * DD;
  float ss = 0.f;
  for (int d = threadIdx.x; d < DD; d += 256) {
    float v = xr[d];
    ss += v * v;
  }
  for (int off = 16; off > 0; off >>= 1) ss += __shfl_down(ss, off, 32);
  if ((threadIdx.x & 31) == 0) red[threadIdx.x >> 5] = ss;
  __syncthreads();
  if (threadIdx.x == 0) {
    float s = 0.f;
    for (int i = 0; i < 8; ++i) s += red[i];
    red[0] = rsqrtf(s + 1e-5f) * 32.0f;  // sqrt(1024) = 32
  }
  __syncthreads();
  const float inv = red[0];
  bf16* xo = xn + (size_t)row * DD;
  for (int d = threadIdx.x; d < DD; d += 256) {
    xo[d] = (bf16)(xr[d] * inv * gamma[d]);
  }
}

// ---------------------------------------------------------------------------
// Kernel 2: pack weights -> bf16, TRANSPOSED.  wcatT[c][d] = Wconcat[d][c],
// c in [q|k|v|a(2D)|g] order; woT[c][d] = wo[d][c].  B^T layout makes both
// GEMM tiles straight 16B contiguous copies (async-to-LDS friendly).
// ---------------------------------------------------------------------------
__global__ __launch_bounds__(256) void gl_pack_weights_kernel(
    const float* __restrict__ wq, const float* __restrict__ wk,
    const float* __restrict__ wv, const float* __restrict__ wa,
    const float* __restrict__ wg, const float* __restrict__ wo,
    bf16* __restrict__ wcatT, bf16* __restrict__ woT) {
  const size_t i = (size_t)blockIdx.x * 256 + threadIdx.x;
  if (i < (size_t)DD * NC6) {
    const int d = (int)(i & (DD - 1));
    const int c = (int)(i >> 10);
    float v;
    if (c < 1024)       v = wq[(size_t)d * DD + c];
    else if (c < 2048)  v = wk[(size_t)d * DD + (c - 1024)];
    else if (c < 3072)  v = wv[(size_t)d * DD + (c - 2048)];
    else if (c < 5120)  v = wa[(size_t)d * (2 * DD) + (c - 3072)];
    else                v = wg[(size_t)d * DD + (c - 5120)];
    wcatT[i] = (bf16)v;
  }
  if (i < (size_t)DD * DD) {
    const int d = (int)(i & (DD - 1));
    const int c = (int)(i >> 10);
    woT[i] = (bf16)wo[(size_t)d * DD + c];
  }
}

// ---------------------------------------------------------------------------
// Kernel 3/7: bf16 WMMA GEMM.  C[M,N] = A[M,K] @ B[K,N] where B is given
// TRANSPOSED (BT[N][K]).  Block tile 128x128 (8 wave32 waves, 2x4), wave tile
// 64x32 (4x2 v_wmma_f32_16x16x32_bf16).  Double-buffered LDS fed by
// global_load_async_to_lds_b128 (ASYNCcnt) when available.
// ---------------------------------------------------------------------------
#define TM 128
#define TN 128
#define TK 32

__global__ __launch_bounds__(256) void gl_gemm_bf16_kernel(
    const bf16* __restrict__ A, const bf16* __restrict__ BT,
    float* __restrict__ C, int M, int N, int K) {
  __shared__ __align__(16) bf16 sA[2][TM][TK];   // 2 x 8 KB
  __shared__ __align__(16) bf16 sB[2][TN][TK];   // 2 x 8 KB (N-major = B^T)

  const int tid  = threadIdx.x;
  const int lane = tid & 31;
  const int wave = tid >> 5;
  const int wm   = wave >> 2;  // 0..1  (M dir)
  const int wn   = wave & 3;   // 0..3  (N dir)
  const int block_m = blockIdx.y * TM;
  const int block_n = blockIdx.x * TN;

  // Tile staging geometry: 128x32 bf16 = 512 x 16B chunks; thread handles
  // chunks {tid, tid+256} -> rows {tr, tr+64}, col tc, for both A and B^T.
  const int tr = tid >> 2;         // 0..63
  const int tc = (tid & 3) * 8;    // 0,8,16,24
  const bf16* Abase = A  + (size_t)(block_m + tr) * K + tc;
  const bf16* Bbase = BT + (size_t)(block_n + tr) * K + tc;

  auto stage = [&](int k0, int buf) {
#if GL_ASYNC
    __builtin_amdgcn_global_load_async_to_lds_b128(
        (gl_gv4i*)(Abase + k0),                  (gl_lv4i*)&sA[buf][tr][tc], 0, 0);
    __builtin_amdgcn_global_load_async_to_lds_b128(
        (gl_gv4i*)(Abase + (size_t)64 * K + k0), (gl_lv4i*)&sA[buf][tr + 64][tc], 0, 0);
    __builtin_amdgcn_global_load_async_to_lds_b128(
        (gl_gv4i*)(Bbase + k0),                  (gl_lv4i*)&sB[buf][tr][tc], 0, 0);
    __builtin_amdgcn_global_load_async_to_lds_b128(
        (gl_gv4i*)(Bbase + (size_t)64 * K + k0), (gl_lv4i*)&sB[buf][tr + 64][tc], 0, 0);
#else
    bf16x8 va0 = *(const bf16x8*)(Abase + k0);
    bf16x8 va1 = *(const bf16x8*)(Abase + (size_t)64 * K + k0);
    bf16x8 vb0 = *(const bf16x8*)(Bbase + k0);
    bf16x8 vb1 = *(const bf16x8*)(Bbase + (size_t)64 * K + k0);
    *(bf16x8*)&sA[buf][tr][tc]      = va0;
    *(bf16x8*)&sA[buf][tr + 64][tc] = va1;
    *(bf16x8*)&sB[buf][tr][tc]      = vb0;
    *(bf16x8*)&sB[buf][tr + 64][tc] = vb1;
#endif
  };

  v8f acc[4][2];
#pragma unroll
  for (int i = 0; i < 4; ++i)
#pragma unroll
    for (int j = 0; j < 2; ++j) {
      v8f z = {0.f, 0.f, 0.f, 0.f, 0.f, 0.f, 0.f, 0.f};
      acc[i][j] = z;
    }

  const int lhalf = lane >> 4;   // 0: lanes 0-15, 1: lanes 16-31
  const int lrow  = lane & 15;
  const int NK    = K / TK;

  stage(0, 0);
#if GL_ASYNC
  gl_wait_async();
#endif
  __syncthreads();

  int buf = 0;
  for (int kt = 0; kt < NK; ++kt) {
    const bool more = (kt + 1 < NK);
    if (more) stage((kt + 1) * TK, buf ^ 1);

    // --- fragments from LDS (ISA 7.12.2 layouts) ---
    // A 16x32 bf16: lanes 0-15 (row M=lrow) hold K {0..7,16..23};
    //               lanes 16-31 hold K {8..15,24..31}.
    v16bf afrag[4];
#pragma unroll
    for (int i = 0; i < 4; ++i) {
      const int m  = wm * 64 + i * 16 + lrow;
      const int kb = lhalf * 8;
      bf16x8 lo = *(const bf16x8*)&sA[buf][m][kb];
      bf16x8 hi = *(const bf16x8*)&sA[buf][m][kb + 16];
      afrag[i] = __builtin_shufflevector(lo, hi, 0, 1, 2, 3, 4, 5, 6, 7,
                                         8, 9, 10, 11, 12, 13, 14, 15);
    }
    // B 32x16 bf16: lanes 0-15 (col N=lrow) hold K {0..7,16..23}; lanes 16-31 K {8..15,24..31}.
    v16bf bfrag[2];
#pragma unroll
    for (int j = 0; j < 2; ++j) {
      const int n  = wn * 32 + j * 16 + lrow;
      const int kb = lhalf * 8;
      bf16x8 lo = *(const bf16x8*)&sB[buf][n][kb];
      bf16x8 hi = *(const bf16x8*)&sB[buf][n][kb + 16];
      bfrag[j] = __builtin_shufflevector(lo, hi, 0, 1, 2, 3, 4, 5, 6, 7,
                                         8, 9, 10, 11, 12, 13, 14, 15);
    }
#pragma unroll
    for (int i = 0; i < 4; ++i)
#pragma unroll
      for (int j = 0; j < 2; ++j)
        acc[i][j] = __builtin_amdgcn_wmma_f32_16x16x32_bf16(
            false, afrag[i], false, bfrag[j], (short)0, acc[i][j], false, false);

#if GL_ASYNC
    if (more) gl_wait_async();
#endif
    __syncthreads();
    buf ^= 1;
  }

  // --- epilogue: C/D layout: VGPR e -> M = e (+8 for upper lane half), N = lane&15
#pragma unroll
  for (int i = 0; i < 4; ++i)
#pragma unroll
    for (int j = 0; j < 2; ++j) {
      const int n = block_n + wn * 32 + j * 16 + lrow;
#pragma unroll
      for (int e = 0; e < 8; ++e) {
        const int m = block_m + wm * 64 + i * 16 + e + lhalf * 8;
        C[(size_t)m * N + n] = acc[i][j][e];
      }
    }
}

// ---------------------------------------------------------------------------
// Gated complex recurrence, 3-pass chunked scan over N (chunk=64).
// P layout: [BN][6144]; cols: q@0, k@1024, v@2048, a_re@3072, a_im@4096, g@5120
// ---------------------------------------------------------------------------
__device__ __forceinline__ void gl_gate(float ar, float ai, float& gr, float& gi) {
  const float r   = sqrtf(ar * ar + ai * ai);
  const float mag = 1.0f / (1.0f + __expf(-r));   // sigmoid(|a|)
  const float sc  = mag / fmaxf(r, 1e-30f);       // mag * (cos,sin)(phase)
  gr = ar * sc;
  gi = ai * sc;
}

__global__ __launch_bounds__(256) void gl_scan_pass1_kernel(
    const float* __restrict__ P, float4* __restrict__ chunkAH) {
  const int tid = blockIdx.x * 256 + threadIdx.x;  // B*NCHUNK*D = 262144
  const int d  = tid & (DD - 1);
  const int bc = tid >> 10;        // b*64 + chunk
  const int rowbase = bc << 6;     // b*4096 + chunk*64
  float Ar = 1.f, Ai = 0.f, hr = 0.f, hi = 0.f;
  for (int t = 0; t < CHUNK; ++t) {
    const size_t base = (size_t)(rowbase + t) * NC6;
    float gr, gi;
    gl_gate(P[base + 3072 + d], P[base + 4096 + d], gr, gi);
    const float kv = P[base + 1024 + d] * P[base + 2048 + d];
    const float nhr = gr * hr - gi * hi + kv;
    const float nhi = gr * hi + gi * hr;
    hr = nhr; hi = nhi;
    const float nAr = gr * Ar - gi * Ai;
    const float nAi = gr * Ai + gi * Ar;
    Ar = nAr; Ai = nAi;
  }
  chunkAH[tid] = make_float4(Ar, Ai, hr, hi);
}

__global__ __launch_bounds__(256) void gl_scan_pass2_kernel(
    const float4* __restrict__ chunkAH, float2* __restrict__ carry) {
  const int tid = blockIdx.x * 256 + threadIdx.x;  // B*D = 4096
  const int d = tid & (DD - 1);
  const int b = tid >> 10;
  float cr = 0.f, ci = 0.f;
  for (int c = 0; c < NCHUNK; ++c) {
    const int idx = ((b * NCHUNK + c) << 10) | d;
    carry[idx] = make_float2(cr, ci);
    const float4 AH = chunkAH[idx];
    const float ncr = AH.x * cr - AH.y * ci + AH.z;
    const float nci = AH.x * ci + AH.y * cr + AH.w;
    cr = ncr; ci = nci;
  }
}

__global__ __launch_bounds__(256) void gl_scan_pass3_kernel(
    const float* __restrict__ P, const float2* __restrict__ carry,
    bf16* __restrict__ ymod) {
  const int tid = blockIdx.x * 256 + threadIdx.x;  // B*NCHUNK*D
  const int d  = tid & (DD - 1);
  const int bc = tid >> 10;
  const int rowbase = bc << 6;
  const float2 c0 = carry[tid];
  float hr = c0.x, hi = c0.y;
  for (int t = 0; t < CHUNK; ++t) {
    const size_t base = (size_t)(rowbase + t) * NC6;
    float gr, gi;
    gl_gate(P[base + 3072 + d], P[base + 4096 + d], gr, gi);
    const float kv = P[base + 1024 + d] * P[base + 2048 + d];
    const float nhr = gr * hr - gi * hi + kv;
    const float nhi = gr * hi + gi * hr;
    hr = nhr; hi = nhi;
    const float q = P[base + d];
    const float g = P[base + 5120 + d];
    const float silu = g / (1.0f + __expf(-g));
    ymod[(size_t)(rowbase + t) * DD + d] = (bf16)(q * hr * silu);
  }
}

// ---------------------------------------------------------------------------
// Launch
// ---------------------------------------------------------------------------
extern "C" void kernel_launch(void* const* d_in, const int* in_sizes, int n_in,
                              void* d_out, int out_size, void* d_ws, size_t ws_size,
                              hipStream_t stream) {
  const float* x     = (const float*)d_in[0];
  const float* gamma = (const float*)d_in[1];
  const float* wq    = (const float*)d_in[2];
  const float* wk    = (const float*)d_in[3];
  const float* wv    = (const float*)d_in[4];
  const float* wa    = (const float*)d_in[5];
  const float* wg    = (const float*)d_in[6];
  const float* wo    = (const float*)d_in[7];
  float* out = (float*)d_out;

  char* ws = (char*)d_ws;
  bf16*   xn      = (bf16*)(ws);                       // 32 MB
  bf16*   wcatT   = (bf16*)(ws + (32ull  << 20));      // 12 MB
  bf16*   woT     = (bf16*)(ws + (44ull  << 20));      //  2 MB
  float*  P       = (float*)(ws + (48ull << 20));      // 384 MB
  float4* chunkAH = (float4*)(ws + (432ull << 20));    //  4 MB
  float2* carry   = (float2*)(ws + (436ull << 20));    //  2 MB
  bf16*   ymod    = (bf16*)(ws + (440ull << 20));      // 32 MB

  gl_rmsnorm_kernel<<<BN, 256, 0, stream>>>(x, gamma, xn);
  gl_pack_weights_kernel<<<(DD * NC6) / 256, 256, 0, stream>>>(
      wq, wk, wv, wa, wg, wo, wcatT, woT);
  gl_gemm_bf16_kernel<<<dim3(NC6 / TN, BN / TM), 256, 0, stream>>>(
      xn, wcatT, P, BN, NC6, DD);
  gl_scan_pass1_kernel<<<(BB * NCHUNK * DD) / 256, 256, 0, stream>>>(P, chunkAH);
  gl_scan_pass2_kernel<<<(BB * DD) / 256, 256, 0, stream>>>(chunkAH, carry);
  gl_scan_pass3_kernel<<<(BB * NCHUNK * DD) / 256, 256, 0, stream>>>(P, carry, ymod);
  gl_gemm_bf16_kernel<<<dim3(DD / TN, BN / TM), 256, 0, stream>>>(
      ymod, woT, out, BN, DD, DD);
}
